// IndexGroupedFCN_46514495815904
// MI455X (gfx1250) — matrix operand order repaired and checked by reading, not compiled, not run on hardware
//
#include <hip/hip_runtime.h>
#include <hip/hip_bf16.h>

// Problem constants (from reference)
#define B_ROWS 1024
#define IN_F   20000
#define NG     64
#define GIN    512
#define H1_D   256
#define GS     128
#define OUTW   (NG * GS)   // 8192

#define LDH    136          // padded LDS k-stride in bf16 elems (mult of 8 -> 16B rows)

typedef __attribute__((ext_vector_type(2)))  __bf16 v2bf;
typedef __attribute__((ext_vector_type(8)))  __bf16 v8bf;
typedef __attribute__((ext_vector_type(16))) __bf16 v16bf;
typedef __attribute__((ext_vector_type(8)))  float  v8f;
typedef __attribute__((ext_vector_type(2)))  float  v2f;

// Build a 16-elem bf16 fragment from two 8-elem (16B) LDS runs: pure register concat.
__device__ __forceinline__ v16bf frag16(const __bf16* p0, const __bf16* p1) {
  v8bf a = *(const v8bf*)p0;
  v8bf b = *(const v8bf*)p1;
  return __builtin_shufflevector(a, b, 0, 1, 2, 3, 4, 5, 6, 7, 8, 9, 10, 11, 12, 13, 14, 15);
}

// Split two f32 into bf16 hi + lo-correction pairs and store packed (one b32 each).
__device__ __forceinline__ void split2(float f0, float f1, __bf16* hi, __bf16* lo) {
  __bf16 h0 = (__bf16)f0, h1 = (__bf16)f1;
  v2bf hp = {h0, h1};
  v2bf lp = {(__bf16)(f0 - (float)h0), (__bf16)(f1 - (float)h1)};
  *(v2bf*)hi = hp;
  *(v2bf*)lo = lp;
}

// bf16x3: c += a_hi*b_hi + a_hi*b_lo + a_lo*b_hi  (f32 accumulate in WMMA)
__device__ __forceinline__ v8f wmma3(v16bf ahi, v16bf alo, v16bf bhi, v16bf blo, v8f c) {
  c = __builtin_amdgcn_wmma_f32_16x16x32_bf16(false, ahi, false, bhi, (short)0, c, false, false);
  c = __builtin_amdgcn_wmma_f32_16x16x32_bf16(false, ahi, false, blo, (short)0, c, false, false);
  c = __builtin_amdgcn_wmma_f32_16x16x32_bf16(false, alo, false, bhi, (short)0, c, false, false);
  return c;
}

// ---------------------------------------------------------------------------
// Kernel 1: WfT_{hi,lo}[g][n][i] = split(W1[g] @ W2[g])^T   (bf16 planes)
// grid (4, 64): 128-row M blocks x group. 8 waves, each a 32(M)x64(N) block.
// ---------------------------------------------------------------------------
__global__ __launch_bounds__(256) void w1w2_kernel(const float* __restrict__ W1,
                                                   const float* __restrict__ W2,
                                                   __bf16* __restrict__ WfT_hi,
                                                   __bf16* __restrict__ WfT_lo) {
  extern __shared__ __bf16 smem[];
  __bf16* ldsAhi = smem;                   // [128 rows i][LDH]
  __bf16* ldsAlo = smem + 128 * LDH;
  __bf16* ldsBhi = smem + 2 * 128 * LDH;   // [128 cols n][LDH] (k-major rows)
  __bf16* ldsBlo = smem + 3 * 128 * LDH;

  const int g  = blockIdx.y;
  const int m0 = blockIdx.x * 128;
  const int t  = threadIdx.x;
  const int lane = t & 31, wid = t >> 5;
  const int lm = lane & 15, khalf = lane >> 4;
  const int mhalf = wid >> 1;   // 0..3 -> 32-row band
  const int nhalf = wid & 1;    // 0..1 -> 64-col band

  v8f acc[2][4];
#pragma unroll
  for (int mi = 0; mi < 2; ++mi)
#pragma unroll
    for (int ni = 0; ni < 4; ++ni)
#pragma unroll
      for (int v = 0; v < 8; ++v) acc[mi][ni][v] = 0.0f;

  const float* w1g = W1 + (size_t)g * GIN * H1_D;
  const float* w2g = W2 + (size_t)g * H1_D * GS;

  for (int kc = 0; kc < 2; ++kc) {
    const int k0 = kc * 128;
    // stage A: W1[m0+r][k0+2kp..+1] -> split planes (coalesced b64 reads)
#pragma unroll
    for (int i = 0; i < 32; ++i) {
      int linear = t + i * 256;
      int kp = linear & 63, r = linear >> 6;
      v2f f = *(const v2f*)(w1g + (size_t)(m0 + r) * H1_D + k0 + 2 * kp);
      split2(f[0], f[1], ldsAhi + r * LDH + 2 * kp, ldsAlo + r * LDH + 2 * kp);
    }
    // stage B transposed: W2[k0+2kp..+1][n] -> split planes (coalesced over n)
#pragma unroll
    for (int i = 0; i < 32; ++i) {
      int linear = t + i * 256;
      int n = linear & 127, kp = linear >> 7;
      float f0 = w2g[(size_t)(k0 + 2 * kp) * GS + n];
      float f1 = w2g[(size_t)(k0 + 2 * kp + 1) * GS + n];
      split2(f0, f1, ldsBhi + n * LDH + 2 * kp, ldsBlo + n * LDH + 2 * kp);
    }
    __syncthreads();

#pragma unroll
    for (int ks = 0; ks < 4; ++ks) {
      const int kb = ks * 32;
      v16bf ah[2], al[2];
#pragma unroll
      for (int mi = 0; mi < 2; ++mi) {
        const __bf16* rh = ldsAhi + (mhalf * 32 + mi * 16 + lm) * LDH;
        const __bf16* rl = ldsAlo + (mhalf * 32 + mi * 16 + lm) * LDH;
        ah[mi] = frag16(rh + kb + khalf * 8, rh + kb + 16 + khalf * 8);
        al[mi] = frag16(rl + kb + khalf * 8, rl + kb + 16 + khalf * 8);
      }
#pragma unroll
      for (int ni = 0; ni < 4; ++ni) {
        const __bf16* bh = ldsBhi + (nhalf * 64 + ni * 16 + lm) * LDH + kb + khalf * 8;
        const __bf16* bl = ldsBlo + (nhalf * 64 + ni * 16 + lm) * LDH + kb + khalf * 8;
        v16bf bhv = frag16(bh, bh + 16);
        v16bf blv = frag16(bl, bl + 16);
#pragma unroll
        for (int mi = 0; mi < 2; ++mi)
          acc[mi][ni] = wmma3(ah[mi], al[mi], bhv, blv, acc[mi][ni]);
      }
    }
    __syncthreads();
  }

  // Store transposed as split bf16 planes: [g][n][i], 8 consecutive i per lane (b128)
#pragma unroll
  for (int mi = 0; mi < 2; ++mi) {
    const int irow = m0 + mhalf * 32 + mi * 16 + khalf * 8;
#pragma unroll
    for (int ni = 0; ni < 4; ++ni) {
      const int n = nhalf * 64 + ni * 16 + lm;
      v8bf hv, lv;
#pragma unroll
      for (int v = 0; v < 8; ++v) {
        float f = acc[mi][ni][v];
        __bf16 h = (__bf16)f;
        hv[v] = h;
        lv[v] = (__bf16)(f - (float)h);
      }
      size_t off = ((size_t)g * GS + n) * GIN + irow;
      *(v8bf*)(WfT_hi + off) = hv;
      *(v8bf*)(WfT_lo + off) = lv;
    }
  }
}

// ---------------------------------------------------------------------------
// Kernel 2: bfuse[g] = b1[g] @ W2[g] + b2[g]   (tiny: 64x128 dots of len 256)
// ---------------------------------------------------------------------------
__global__ void biasfuse_kernel(const float* __restrict__ b1,
                                const float* __restrict__ W2,
                                const float* __restrict__ b2,
                                float* __restrict__ bf) {
  const int g = blockIdx.x, n = threadIdx.x;
  float a = b2[g * GS + n];
  const float* w2g = W2 + (size_t)g * H1_D * GS;
  for (int h = 0; h < H1_D; ++h) a += b1[g * H1_D + h] * w2g[(size_t)h * GS + n];
  bf[g * GS + n] = a;
}

// ---------------------------------------------------------------------------
// Kernel 3 (main): out[b, g*128+n] = sum_i x[b, idx[g,i]] * Wf[g,i,n] + bf[g,n]
// grid (8, 64): 128-batch block x group. K=512 in four 128-chunks.
// A: gather+split into LDS planes.  B: pure b128 copy of prebuilt bf16 planes.
// ---------------------------------------------------------------------------
__global__ __launch_bounds__(256) void fused_kernel(const float* __restrict__ x,
                                                    const int* __restrict__ idx,
                                                    const __bf16* __restrict__ WfT_hi,
                                                    const __bf16* __restrict__ WfT_lo,
                                                    const float* __restrict__ bf,
                                                    float* __restrict__ out) {
  extern __shared__ __bf16 smem[];
  __bf16* ldsAhi = smem;                   // [128 rows b][LDH]
  __bf16* ldsAlo = smem + 128 * LDH;
  __bf16* ldsBhi = smem + 2 * 128 * LDH;   // [128 cols n][LDH]
  __bf16* ldsBlo = smem + 3 * 128 * LDH;
  int*    ldsI   = (int*)(smem + 4 * 128 * LDH);   // [128] idx chunk

  const int g  = blockIdx.y;
  const int b0 = blockIdx.x * 128;
  const int t  = threadIdx.x;
  const int lane = t & 31, wid = t >> 5;
  const int lm = lane & 15, khalf = lane >> 4;
  const int mhalf = wid >> 1;   // 0..3
  const int nhalf = wid & 1;    // 0..1

  // accumulators initialized with the fused bias (bias depends only on N)
  v8f acc[2][4];
#pragma unroll
  for (int ni = 0; ni < 4; ++ni) {
    float bv = bf[g * GS + nhalf * 64 + ni * 16 + lm];
#pragma unroll
    for (int mi = 0; mi < 2; ++mi)
#pragma unroll
      for (int v = 0; v < 8; ++v) acc[mi][ni][v] = bv;
  }

  const __bf16* bhsrc = WfT_hi + (size_t)g * GS * GIN;
  const __bf16* blsrc = WfT_lo + (size_t)g * GS * GIN;

  for (int kc = 0; kc < 4; ++kc) {
    const int k0 = kc * 128;
    if (t < 128) ldsI[t] = idx[g * GIN + k0 + t];
    __syncthreads();

    // gather A: x[b0+r, idx[g,k0+2kp..+1]] -> split planes
#pragma unroll
    for (int i = 0; i < 32; ++i) {
      int linear = t + i * 256;
      int kp = linear & 63, r = linear >> 6;
      const float* xr = x + (size_t)(b0 + r) * IN_F;
      float f0 = xr[ldsI[2 * kp]];
      float f1 = xr[ldsI[2 * kp + 1]];
      split2(f0, f1, ldsAhi + r * LDH + 2 * kp, ldsAlo + r * LDH + 2 * kp);
    }
    // copy B: prebuilt bf16 planes, b128 both sides, no conversion
#pragma unroll
    for (int i = 0; i < 8; ++i) {
      int linear = t + i * 256;          // 0..2047 chunks of 8 bf16
      int n = linear >> 4, c = (linear & 15) * 8;
      *(v8bf*)(ldsBhi + n * LDH + c) = *(const v8bf*)(bhsrc + (size_t)n * GIN + k0 + c);
      *(v8bf*)(ldsBlo + n * LDH + c) = *(const v8bf*)(blsrc + (size_t)n * GIN + k0 + c);
    }
    if (kc < 3) __builtin_prefetch(bhsrc + (size_t)(t >> 1) * GIN + k0 + 128 + (t & 1) * 64, 0, 1);
    __syncthreads();

#pragma unroll
    for (int ks = 0; ks < 4; ++ks) {
      const int kb = ks * 32;
      v16bf ah[2], al[2];
#pragma unroll
      for (int mi = 0; mi < 2; ++mi) {
        const __bf16* rh = ldsAhi + (mhalf * 32 + mi * 16 + lm) * LDH;
        const __bf16* rl = ldsAlo + (mhalf * 32 + mi * 16 + lm) * LDH;
        ah[mi] = frag16(rh + kb + khalf * 8, rh + kb + 16 + khalf * 8);
        al[mi] = frag16(rl + kb + khalf * 8, rl + kb + 16 + khalf * 8);
      }
#pragma unroll
      for (int ni = 0; ni < 4; ++ni) {
        const __bf16* bh = ldsBhi + (nhalf * 64 + ni * 16 + lm) * LDH + kb + khalf * 8;
        const __bf16* bl = ldsBlo + (nhalf * 64 + ni * 16 + lm) * LDH + kb + khalf * 8;
        v16bf bhv = frag16(bh, bh + 16);
        v16bf blv = frag16(bl, bl + 16);
#pragma unroll
        for (int mi = 0; mi < 2; ++mi)
          acc[mi][ni] = wmma3(ah[mi], al[mi], bhv, blv, acc[mi][ni]);
      }
    }
    __syncthreads();
  }

  // C/D layout: lanes 0-15 -> M=v, lanes 16-31 -> M=v+8; N = lane&15
#pragma unroll
  for (int mi = 0; mi < 2; ++mi) {
    const int mtop = b0 + mhalf * 32 + mi * 16 + khalf * 8;
#pragma unroll
    for (int ni = 0; ni < 4; ++ni) {
      const int col = g * GS + nhalf * 64 + ni * 16 + lm;
#pragma unroll
      for (int v = 0; v < 8; ++v)
        out[(size_t)(mtop + v) * OUTW + col] = acc[mi][ni][v];
    }
  }
}

// ---------------------------------------------------------------------------
extern "C" void kernel_launch(void* const* d_in, const int* in_sizes, int n_in,
                              void* d_out, int out_size, void* d_ws, size_t ws_size,
                              hipStream_t stream) {
  const float* x   = (const float*)d_in[0];
  const int*   idx = (const int*)d_in[1];
  const float* W1  = (const float*)d_in[2];
  const float* b1  = (const float*)d_in[3];
  const float* W2  = (const float*)d_in[4];
  const float* b2  = (const float*)d_in[5];
  float* out = (float*)d_out;

  __bf16* WfT_hi = (__bf16*)d_ws;                         // 64*128*512 bf16 = 8 MB
  __bf16* WfT_lo = WfT_hi + (size_t)NG * GS * GIN;        // 8 MB
  float*  bf     = (float*)(WfT_lo + (size_t)NG * GS * GIN);

  const size_t smem_gemm  = (size_t)(4 * 128 * LDH) * sizeof(__bf16);
  const size_t smem_fused = smem_gemm + 128 * sizeof(int);

  w1w2_kernel<<<dim3(4, NG), 256, smem_gemm, stream>>>(W1, W2, WfT_hi, WfT_lo);
  biasfuse_kernel<<<NG, GS, 0, stream>>>(b1, W2, b2, bf);
  fused_kernel<<<dim3(B_ROWS / 128, NG), 256, smem_fused, stream>>>(x, idx, WfT_hi, WfT_lo, bf, out);
}
